// FullAttention_13511967113470
// MI455X (gfx1250) — matrix operand order repaired and checked by reading, not compiled, not run on hardware
//
#include <hip/hip_runtime.h>
#include <hip/hip_bf16.h>
#include <cstdint>

#define BB   2
#define NN   1536
#define DIMV 1536
#define HH   8
#define HD   512      // H*DK
#define N2   3072     // 2N (padded rel rows; real rows = 2N-1)
#define NREL 3071     // 2N-1

typedef __attribute__((ext_vector_type(16))) __bf16 v16bf;
typedef __attribute__((ext_vector_type(8)))  float  v8f;
typedef unsigned int v4u  __attribute__((ext_vector_type(4)));
typedef int          v8i_t __attribute__((ext_vector_type(8)));
typedef int          v4i_t __attribute__((ext_vector_type(4)));

#if __has_builtin(__builtin_amdgcn_tensor_load_to_lds) && \
    __has_builtin(__builtin_amdgcn_s_wait_tensorcnt)
#define HAVE_TDM 1
#else
#define HAVE_TDM 0
#endif

union FragBF { uint32_t u[8]; v16bf v; };

__device__ __forceinline__ uint16_t f2bf(float f) {
  uint32_t u = __float_as_uint(f);
  uint32_t r = u + 0x7FFFu + ((u >> 16) & 1u);   // round-to-nearest-even
  return (uint16_t)(r >> 16);
}

__device__ __forceinline__ float bperm_f(int srcLane, float x) {
  return __int_as_float(__builtin_amdgcn_ds_bpermute(srcLane << 2, __float_as_int(x)));
}

#if HAVE_TDM
// TDM: load a [tileRows x 32] bf16 tile (row stride K elems) from `gptr`
// into LDS offset `lds_off`, padded to a 40-element LDS row stride
// (pad_interval=3 -> every 16 DWORDs, pad_amount=3 -> 4 DWORDs).
// Rows beyond `validRows` are zero-filled by TDM OOB handling.
__device__ __forceinline__ void tdm_load_a32(uint32_t lds_off, const uint16_t* gptr,
                                             uint32_t validRows, uint32_t K) {
  uint64_t ga = (uint64_t)(uintptr_t)gptr;
  v4u g0;
  g0[0] = 1u;                                   // count=1, user mode
  g0[1] = lds_off;                              // lds_addr (bytes)
  g0[2] = (uint32_t)(ga & 0xFFFFFFFFu);         // global_addr[31:0]
  g0[3] = (uint32_t)((ga >> 32) & 0x01FFFFFFu)  // global_addr[56:32]
          | (2u << 30);                         // type = 2 ("image")
  v8i_t g1;
  g1[0] = (int)((1u << 16)                      // data_size = 2 bytes
                | (1u << 20)                    // pad_enable
                | (3u << 22)                    // pad_interval: 16 DWORDs
                | (3u << 25));                  // pad_amount:   4 DWORDs
  g1[1] = (int)(32u << 16);                     // tensor_dim0 = 32 (bits 79:48 lo)
  g1[2] = (int)((validRows & 0xFFFFu) << 16);   // tensor_dim0 hi=0 | tensor_dim1 lo
  g1[3] = (int)(((validRows >> 16) & 0xFFFFu)   // tensor_dim1 hi
                | (32u << 16));                 // tile_dim0 = 32
  g1[4] = (int)128u;                            // tile_dim1 = 128, tile_dim2 = 0
  g1[5] = (int)K;                               // tensor_dim0_stride lo
  g1[6] = 0;                                    // stride hi | dim1_stride lo
  g1[7] = 0;
  v4i_t gz4 = {0, 0, 0, 0};
#if defined(__clang_major__) && (__clang_major__ >= 23)
  v8i_t gz8 = {0, 0, 0, 0, 0, 0, 0, 0};
  __builtin_amdgcn_tensor_load_to_lds(g0, g1, gz4, gz4, gz8, 0);
#else
  __builtin_amdgcn_tensor_load_to_lds(g0, g1, gz4, gz4, 0);
#endif
}
#endif

// ---------------------------------------------------------------- converts
__global__ void cvt_f32_bf16(const float* __restrict__ src,
                             uint16_t* __restrict__ dst, int n) {
  int i = blockIdx.x * blockDim.x + threadIdx.x;
  if (i < n) dst[i] = f2bf(src[i]);
}

// ---------------------------------------------------- generic bf16 GEMM
// C[M,Nc] (f32) = A[M,K](bf16) x B[K,Nc](bf16) (+ bias[Nc] if non-null)
// block tile 128x64, 256 threads = 8 waves, wave tile 32x32 (2x2 WMMA)
__global__ __launch_bounds__(256)
void gemm_bf16(const uint16_t* __restrict__ A, const uint16_t* __restrict__ B,
               float* __restrict__ C, const float* __restrict__ bias,
               int M, int K, int Nc) {
  // single allocation so ldsA sits at LDS offset 0 (TDM descriptor target)
  __shared__ uint16_t lds[128 * 40 + 64 * 40];
  uint16_t* ldsA  = lds;              // [128][40] row-major, 16B-aligned rows
  uint16_t* ldsBT = lds + 128 * 40;   // transposed: [n][k], stride 40

  const int tid = threadIdx.x;
  const int lane = tid & 31;
  const int w = tid >> 5;
  const int wm = w >> 1, wn = w & 1;
  const int m0 = blockIdx.y * 128;
  const int n0 = blockIdx.x * 64;
  const int laneHi = (lane >> 4) & 1;
  const int lmod = lane & 15;

  v8f acc[2][2] = {};

  for (int kk = 0; kk < K; kk += 32) {
#if HAVE_TDM
    // stage A tile 128x32 with the Tensor Data Mover (wave 0 only; TDM
    // ignores EXEC, zero-fills rows beyond M, and pads LDS rows to 40 elems)
    if (w == 0) {
      tdm_load_a32(0u, A + (size_t)m0 * K + kk, (uint32_t)(M - m0), (uint32_t)K);
      __builtin_amdgcn_s_wait_tensorcnt(0);
    }
#else
    // manual staging fallback
    {
      const int row = tid >> 1, seg = tid & 1;
      uint4 d0 = make_uint4(0, 0, 0, 0), d1 = d0;
      if (m0 + row < M) {
        const uint16_t* ga = A + (size_t)(m0 + row) * K + kk + seg * 16;
        d0 = *(const uint4*)(ga);
        d1 = *(const uint4*)(ga + 8);
      }
      uint4* lp = (uint4*)&ldsA[row * 40 + seg * 16];
      lp[0] = d0; lp[1] = d1;
    }
#endif
    // stage B tile 32x64, transposed into [n][k]
    {
      const int kr = tid >> 3, c = tid & 7;
      const uint16_t* gb = B + (size_t)(kk + kr) * Nc + n0 + c * 8;
      uint4 d = *(const uint4*)gb;
      __builtin_prefetch(gb + 32 * Nc, 0, 0);   // next K-step (global_prefetch_b8)
      const uint16_t* e = (const uint16_t*)&d;
#pragma unroll
      for (int q = 0; q < 8; ++q) ldsBT[(c * 8 + q) * 40 + kr] = e[q];
    }
    __syncthreads();

    FragBF af[2], bfr[2];
#pragma unroll
    for (int mi = 0; mi < 2; ++mi) {
      const int row = wm * 32 + mi * 16 + lmod;
#pragma unroll
      for (int v = 0; v < 8; ++v) {
        const int ka = (v >> 2) * 16 + laneHi * 8 + 2 * (v & 3);   // A 16x32 map
        af[mi].u[v] = *(const uint32_t*)&ldsA[row * 40 + ka];
      }
    }
#pragma unroll
    for (int ni = 0; ni < 2; ++ni) {
      const int col = wn * 32 + ni * 16 + lmod;
#pragma unroll
      for (int v = 0; v < 8; ++v) {
        const int kb = (v >> 2) * 16 + laneHi * 8 + 2 * (v & 3);   // B 32x16 map
        bfr[ni].u[v] = *(const uint32_t*)&ldsBT[col * 40 + kb];
      }
    }
#pragma unroll
    for (int mi = 0; mi < 2; ++mi)
#pragma unroll
      for (int ni = 0; ni < 2; ++ni)
        acc[mi][ni] = __builtin_amdgcn_wmma_f32_16x16x32_bf16(
            false, af[mi].v, false, bfr[ni].v, (short)0, acc[mi][ni], false, false);
    __syncthreads();
  }

  // epilogue: hoist bias (two values per lane), then plain guarded stores
  float bv[2] = {0.0f, 0.0f};
  if (bias) {
#pragma unroll
    for (int ni = 0; ni < 2; ++ni) bv[ni] = bias[n0 + wn * 32 + ni * 16 + lmod];
  }
#pragma unroll
  for (int mi = 0; mi < 2; ++mi)
#pragma unroll
    for (int ni = 0; ni < 2; ++ni)
#pragma unroll
      for (int r = 0; r < 8; ++r) {
        const int row = wm * 32 + mi * 16 + r + laneHi * 8;
        const int col = wn * 32 + ni * 16 + lmod;
        const int gr = m0 + row, gc = n0 + col;
        if (gr < M) C[(size_t)gr * Nc + gc] = acc[mi][ni][r] + bv[ni];
      }
}

// ------------------------------------------------- layout rearrangements
__global__ void rearrange_qkv(const float* __restrict__ q, const float* __restrict__ k,
                              const float* __restrict__ v, const float* __restrict__ cb,
                              const float* __restrict__ rpb,
                              uint16_t* __restrict__ qc, uint16_t* __restrict__ qr,
                              uint16_t* __restrict__ kb, uint16_t* __restrict__ vt) {
  int idx = blockIdx.x * blockDim.x + threadIdx.x;   // over B*N*HD
  if (idx >= BB * NN * HD) return;
  const int row = idx >> 9;           // HD = 512
  const int col = idx & 511;
  const int b = row / NN, n = row % NN;
  const int h = col >> 6, d = col & 63;
  const float scale = 0.125f;         // DK^-0.5
  const float qv = q[idx] * scale;
  const size_t o = ((size_t)(b * HH + h) * NN + n) * 64 + d;
  qc[o] = f2bf(qv + cb[col]);
  qr[o] = f2bf(qv + rpb[col]);
  kb[o] = f2bf(k[idx]);
  vt[((size_t)(b * HH + h) * 64 + d) * NN + n] = f2bf(v[idx]);
}

__global__ void rearrange_relk(const float* __restrict__ rk, uint16_t* __restrict__ out) {
  int idx = blockIdx.x * blockDim.x + threadIdx.x;   // over H*N2*64
  if (idx >= HH * N2 * 64) return;
  const int d = idx & 63;
  const int m = (idx >> 6) % N2;
  const int h = idx / (N2 * 64);
  out[idx] = (m < NREL) ? f2bf(rk[(size_t)m * HD + h * 64 + d]) : (uint16_t)0;
}

// ------------------------------------------------------ flash attention
// one wave (32 threads) per 16-row tile of one (b,h); streams 32 cols/step
__global__ __launch_bounds__(32)
void attn_kernel(const uint16_t* __restrict__ qc, const uint16_t* __restrict__ qr,
                 const uint16_t* __restrict__ kmat, const uint16_t* __restrict__ vt,
                 const uint16_t* __restrict__ relk, uint16_t* __restrict__ ob) {
  __shared__ uint16_t ldsP[16 * 36];

  const int lane = threadIdx.x & 31;
  const int laneHi = (lane >> 4) & 1;
  const int lmod = lane & 15;
  const int i0 = blockIdx.x * 16;
  const int h = blockIdx.y;
  const int b = blockIdx.z;

  const uint16_t* qcb   = qc   + ((size_t)(b * HH + h) * NN) * 64;
  const uint16_t* qrb   = qr   + ((size_t)(b * HH + h) * NN) * 64;
  const uint16_t* kbase = kmat + ((size_t)(b * HH + h) * NN) * 64;
  const uint16_t* vbase = vt   + ((size_t)(b * HH + h) * 64) * NN;
  const uint16_t* rbase = relk + (size_t)h * N2 * 64;

  // Q fragments (A layout), content-biased and pos-biased variants
  FragBF qcf[2], qrf[2];
#pragma unroll
  for (int dh = 0; dh < 2; ++dh)
#pragma unroll
    for (int v = 0; v < 8; ++v) {
      const int ka = (v >> 2) * 16 + laneHi * 8 + 2 * (v & 3);
      qcf[dh].u[v] = *(const uint32_t*)&qcb[(size_t)(i0 + lmod) * 64 + dh * 32 + ka];
      qrf[dh].u[v] = *(const uint32_t*)&qrb[(size_t)(i0 + lmod) * 64 + dh * 32 + ka];
    }

  float mrow[8], lrow[8];
#pragma unroll
  for (int r = 0; r < 8; ++r) { mrow[r] = -1e30f; lrow[r] = 0.0f; }
  v8f accO[4] = {};

  for (int jw = 0; jw < NN; jw += 32) {
    // content logits S = Qc * K^T  (16 x 32)
    v8f S[2] = {};
#pragma unroll
    for (int js = 0; js < 2; ++js)
#pragma unroll
      for (int dh = 0; dh < 2; ++dh) {
        FragBF kf;
#pragma unroll
        for (int v = 0; v < 8; ++v) {
          const int kb = (v >> 2) * 16 + laneHi * 8 + 2 * (v & 3);
          kf.u[v] = *(const uint32_t*)
              &kbase[(size_t)(jw + js * 16 + lmod) * 64 + dh * 32 + kb];
        }
        S[js] = __builtin_amdgcn_wmma_f32_16x16x32_bf16(
            false, qcf[dh].v, false, kf.v, (short)0, S[js], false, false);
      }

    // rel logits window E[ti,u] = Qr . relK[m0+u], u in [0,48)
    const int m0 = (NN - 1) + jw - i0 - 15;
    v8f E[3] = {};
#pragma unroll
    for (int ut = 0; ut < 3; ++ut)
#pragma unroll
      for (int dh = 0; dh < 2; ++dh) {
        FragBF rf;
#pragma unroll
        for (int v = 0; v < 8; ++v) {
          const int kb = (v >> 2) * 16 + laneHi * 8 + 2 * (v & 3);
          rf.u[v] = *(const uint32_t*)
              &rbase[(size_t)(m0 + ut * 16 + lmod) * 64 + dh * 32 + kb];
        }
        E[ut] = __builtin_amdgcn_wmma_f32_16x16x32_bf16(
            false, qrf[dh].v, false, rf.v, (short)0, E[ut], false, false);
      }

    // Transformer-XL shift: S[ti,tj] += E[ti, tj - ti + 15] (banded bpermute)
#pragma unroll
    for (int r = 0; r < 8; ++r) {
      const int ti = r + laneHi * 8;
      {
        const int u = lmod - ti + 15;                 // 0..30
        const int sl = (u & 15) | (lane & 16);
        const float e0 = bperm_f(sl, E[0][r]);
        const float e1 = bperm_f(sl, E[1][r]);
        S[0][r] += (u >= 16) ? e1 : e0;
      }
      {
        const int u = 16 + lmod - ti + 15;            // 16..46
        const int sl = (u & 15) | (lane & 16);
        const float e1 = bperm_f(sl, E[1][r]);
        const float e2 = bperm_f(sl, E[2][r]);
        S[1][r] += (u >= 32) ? e2 : e1;
      }
    }

    // online softmax update per row, write P (bf16) to LDS
#pragma unroll
    for (int r = 0; r < 8; ++r) {
      float mx = fmaxf(S[0][r], S[1][r]);
#pragma unroll
      for (int mk = 8; mk >= 1; mk >>= 1) mx = fmaxf(mx, bperm_f(lane ^ mk, mx));
      const float mnew = fmaxf(mrow[r], mx);
      const float corr = __expf(mrow[r] - mnew);
      const float p0 = __expf(S[0][r] - mnew);
      const float p1 = __expf(S[1][r] - mnew);
      float ps = p0 + p1;
#pragma unroll
      for (int mk = 8; mk >= 1; mk >>= 1) ps += bperm_f(lane ^ mk, ps);
      lrow[r] = lrow[r] * corr + ps;
      mrow[r] = mnew;
#pragma unroll
      for (int nt = 0; nt < 4; ++nt) accO[nt][r] *= corr;
      ldsP[(r + laneHi * 8) * 36 + lmod]      = f2bf(p0);
      ldsP[(r + laneHi * 8) * 36 + 16 + lmod] = f2bf(p1);
    }
    __syncthreads();

    // P as A fragment (16 x 32)
    FragBF pf;
#pragma unroll
    for (int v = 0; v < 8; ++v) {
      const int ka = (v >> 2) * 16 + laneHi * 8 + 2 * (v & 3);
      pf.u[v] = *(const uint32_t*)&ldsP[lmod * 36 + ka];
    }

    // O += P * V   (V pre-transposed [dv][n] so K-pairs are contiguous)
#pragma unroll
    for (int nt = 0; nt < 4; ++nt) {
      FragBF vf;
#pragma unroll
      for (int v = 0; v < 8; ++v) {
        const int kb = (v >> 2) * 16 + laneHi * 8 + 2 * (v & 3);
        vf.u[v] = *(const uint32_t*)&vbase[(size_t)(nt * 16 + lmod) * NN + jw + kb];
      }
      accO[nt] = __builtin_amdgcn_wmma_f32_16x16x32_bf16(
          false, pf.v, false, vf.v, (short)0, accO[nt], false, false);
    }
    __syncthreads();
  }

  // normalize and emit bf16 [B*N, H*DV] for the output projection GEMM
#pragma unroll
  for (int nt = 0; nt < 4; ++nt)
#pragma unroll
    for (int r = 0; r < 8; ++r) {
      const int row = i0 + r + laneHi * 8;
      const int col = h * 64 + nt * 16 + lmod;
      ob[(size_t)(b * NN + row) * HD + col] = f2bf(accO[nt][r] / lrow[r]);
    }
}

// ------------------------------------------------------------------ host
static inline int ceil_div(int a, int b) { return (a + b - 1) / b; }

extern "C" void kernel_launch(void* const* d_in, const int* in_sizes, int n_in,
                              void* d_out, int out_size, void* d_ws, size_t ws_size,
                              hipStream_t stream) {
  (void)in_sizes; (void)n_in; (void)out_size; (void)ws_size;
  const float* x    = (const float*)d_in[0];
  const float* Wq   = (const float*)d_in[1];
  const float* Wk   = (const float*)d_in[2];
  const float* Wv   = (const float*)d_in[3];
  const float* Wrel = (const float*)d_in[4];
  const float* Wout = (const float*)d_in[5];
  const float* bout = (const float*)d_in[6];
  const float* cb   = (const float*)d_in[7];
  const float* rpb  = (const float*)d_in[8];
  const float* pe   = (const float*)d_in[9];

  char* ws = (char*)d_ws;
  size_t off = 0;
  auto alloc = [&](size_t bytes) -> char* {
    char* p = ws + off;
    off += (bytes + 255) & ~(size_t)255;
    return p;
  };

  uint16_t* xb    = (uint16_t*)alloc((size_t)BB * NN * DIMV * 2);
  uint16_t* wqb   = (uint16_t*)alloc((size_t)DIMV * HD * 2);
  uint16_t* wkb   = (uint16_t*)alloc((size_t)DIMV * HD * 2);
  uint16_t* wvb   = (uint16_t*)alloc((size_t)DIMV * HD * 2);
  uint16_t* wrelb = (uint16_t*)alloc((size_t)192 * HD * 2);
  uint16_t* woutb = (uint16_t*)alloc((size_t)HD * DIMV * 2);
  uint16_t* peb   = (uint16_t*)alloc((size_t)NREL * 192 * 2);
  float*    qf    = (float*)alloc((size_t)BB * NN * HD * 4);
  float*    kf    = (float*)alloc((size_t)BB * NN * HD * 4);
  float*    vf    = (float*)alloc((size_t)BB * NN * HD * 4);
  float*    rkf   = (float*)alloc((size_t)NREL * HD * 4);
  uint16_t* qcb   = (uint16_t*)alloc((size_t)BB * HH * NN * 64 * 2);
  uint16_t* qrb   = (uint16_t*)alloc((size_t)BB * HH * NN * 64 * 2);
  uint16_t* kbb   = (uint16_t*)alloc((size_t)BB * HH * NN * 64 * 2);
  uint16_t* vbt   = (uint16_t*)alloc((size_t)BB * HH * 64 * NN * 2);
  uint16_t* rkb   = (uint16_t*)alloc((size_t)HH * N2 * 64 * 2);
  uint16_t* obuf  = (uint16_t*)alloc((size_t)BB * NN * HD * 2);

  const int CT = 256;
  // bf16 conversions
  cvt_f32_bf16<<<ceil_div(BB * NN * DIMV, CT), CT, 0, stream>>>(x, xb, BB * NN * DIMV);
  cvt_f32_bf16<<<ceil_div(DIMV * HD, CT), CT, 0, stream>>>(Wq, wqb, DIMV * HD);
  cvt_f32_bf16<<<ceil_div(DIMV * HD, CT), CT, 0, stream>>>(Wk, wkb, DIMV * HD);
  cvt_f32_bf16<<<ceil_div(DIMV * HD, CT), CT, 0, stream>>>(Wv, wvb, DIMV * HD);
  cvt_f32_bf16<<<ceil_div(192 * HD, CT), CT, 0, stream>>>(Wrel, wrelb, 192 * HD);
  cvt_f32_bf16<<<ceil_div(HD * DIMV, CT), CT, 0, stream>>>(Wout, woutb, HD * DIMV);
  cvt_f32_bf16<<<ceil_div(NREL * 192, CT), CT, 0, stream>>>(pe, peb, NREL * 192);

  // projection GEMMs: [3072,1536] x [1536,512]
  {
    dim3 g(HD / 64, ceil_div(BB * NN, 128));
    gemm_bf16<<<g, 256, 0, stream>>>(xb, wqb, qf, nullptr, BB * NN, DIMV, HD);
    gemm_bf16<<<g, 256, 0, stream>>>(xb, wkb, kf, nullptr, BB * NN, DIMV, HD);
    gemm_bf16<<<g, 256, 0, stream>>>(xb, wvb, vf, nullptr, BB * NN, DIMV, HD);
  }
  // rel_k GEMM: [3071,192] x [192,512] (TDM zero-fills the M edge)
  {
    dim3 g(HD / 64, ceil_div(NREL, 128));
    gemm_bf16<<<g, 256, 0, stream>>>(peb, wrelb, rkf, nullptr, NREL, 192, HD);
  }

  // head-major rearrangement + bias/scale + bf16
  rearrange_qkv<<<ceil_div(BB * NN * HD, CT), CT, 0, stream>>>(
      qf, kf, vf, cb, rpb, qcb, qrb, kbb, vbt);
  rearrange_relk<<<ceil_div(HH * N2 * 64, CT), CT, 0, stream>>>(rkf, rkb);

  // streaming attention with fused relative shift
  {
    dim3 g(NN / 16, HH, BB);
    attn_kernel<<<g, 32, 0, stream>>>(qcb, qrb, kbb, vbt, rkb, obuf);
  }

  // output projection: [3072,512] x [512,1536] + bout
  {
    dim3 g(DIMV / 64, ceil_div(BB * NN, 128));
    gemm_bf16<<<g, 256, 0, stream>>>(obuf, woutb, (float*)d_out, bout,
                                     BB * NN, HD, DIMV);
  }
}
